// LongCatImageAttention_21895743275339
// MI455X (gfx1250) — matrix operand order repaired and checked by reading, not compiled, not run on hardware
//
#include <hip/hip_runtime.h>
#include <hip/hip_bf16.h>
#include <math.h>

// ---------------------------------------------------------------------------
// LongCat image attention for MI455X (gfx1250, wave32, WMMA).
// All heavy math runs on v_wmma_f32_16x16x32_bf16 (f32 accumulate); norms,
// RoPE and softmax stay in fp32. fp32 operands are converted to bf16 (RNE)
// while staging into LDS in WMMA-fragment order. GEMMs are double-buffered
// (1 barrier / K-step); pure bf16 copies use GLOBAL_LOAD_ASYNC_TO_LDS_B128.
// ---------------------------------------------------------------------------

#define TXT   256
#define IMG   2048
#define SEQ   (TXT + IMG)      // 2304
#define DM    2048
#define HEADS 16
#define HD    128
#define INNER (HEADS * HD)     // 2048
#define N3    (3 * INNER)      // 6144
#define EPSV   1e-5f
#define SCALEV 0.08838834764831845f   // 128^-0.5

typedef __attribute__((ext_vector_type(16))) __bf16 bf16x16;
typedef __attribute__((ext_vector_type(8)))  float  f32x8;

// ---- gfx1250 async load-to-LDS path (guarded; falls back to sync copies) --
#if defined(__HIP_DEVICE_COMPILE__) && \
    __has_builtin(__builtin_amdgcn_global_load_async_to_lds_b128) && \
    __has_builtin(__builtin_amdgcn_s_wait_asynccnt)
#define HAVE_ASYNC_LDS 1
typedef int v4i __attribute__((vector_size(16)));      // b128 transfer unit
typedef __attribute__((address_space(1))) v4i* gv4i_p; // global int4*
typedef __attribute__((address_space(3))) v4i* lv4i_p; // LDS int4*
__device__ __forceinline__ void async_ld_b128(const void* g, void* l) {
  __builtin_amdgcn_global_load_async_to_lds_b128((gv4i_p)g, (lv4i_p)l, 0, 0);
}
__device__ __forceinline__ void wait_async0() {
  __builtin_amdgcn_s_wait_asynccnt(0);
}
#else
#define HAVE_ASYNC_LDS 0
__device__ __forceinline__ void wait_async0() {}
#endif

__device__ __forceinline__ unsigned short f2bf(float f) {
  union { float f; unsigned u; } x; x.f = f;
  unsigned u = x.u;
  u += 0x7FFFu + ((u >> 16) & 1u);      // round-to-nearest-even
  return (unsigned short)(u >> 16);
}
__device__ __forceinline__ unsigned pk2(float a, float b) {
  return (unsigned)f2bf(a) | ((unsigned)f2bf(b) << 16);
}
__device__ __forceinline__ f32x8 zero8() {
  f32x8 z = {0.f,0.f,0.f,0.f,0.f,0.f,0.f,0.f}; return z;
}
__device__ __forceinline__ f32x8 wmma_bf16(bf16x16 a, bf16x16 b, f32x8 c) {
  // (neg_a, A, neg_b, B, c_mod, C, reuse_a, reuse_b)
  return __builtin_amdgcn_wmma_f32_16x16x32_bf16(false, a, false, b,
                                                 (short)0, c, false, false);
}
// 32-byte aligned fragment load (LDS or global)
__device__ __forceinline__ bf16x16 ldfrag(const void* p) {
  return *(const bf16x16*)p;
}
// fragment assembled from two 16-byte halves (16B-aligned sources)
__device__ __forceinline__ bf16x16 ldfrag2(const void* p0, const void* p1) {
  struct { uint4 a, b; } t;
  t.a = *(const uint4*)p0;
  t.b = *(const uint4*)p1;
  bf16x16 r;
  __builtin_memcpy(&r, &t, 32);
  return r;
}

// ---------------------------------------------------------------------------
// Generic bf16 GEMM: C[M,N] = A[M,K] * W[K,N] (+ bias), A fp32 or bf16,
// W fp32 (converted while staging). BM=BN=128, BK=32, 256 threads = 8 waves,
// each wave owns a 64x32 patch (4x2 WMMA tiles). Double-buffered LDS tiles
// (2 x 16KB) in WMMA-fragment order: 32 lanes x 16 bf16 contiguous per lane
// per 16x32 A / 32x16 B subtile => one 32B fragment load.
// ---------------------------------------------------------------------------
struct GemmStage {
  float4 a4[4];     // fp32-A path: raw float4s for the next A tile
  uint4  ab[2];     // bf16-A fallback path (no async builtin)
  float  b[16];     // next W tile column (16 k-rows, fixed n)
};

template<bool ABF16>
__device__ __forceinline__ void gemm_stage_load(
    GemmStage& st, const float* Af, const unsigned short* Ab,
    const float* W, int K, int N, int m0, int n0, int kt, int tid,
    unsigned char* abuf)
{
  if (ABF16) {
#if HAVE_ASYNC_LDS
    // pure copy: DMA straight into the fragment-ordered LDS slots
    #pragma unroll
    for (int i = 0; i < 2; ++i) {
      int idx = tid + i * 256;                 // 0..511
      int m = idx >> 2, k0 = (idx & 3) * 8;    // 8 consecutive bf16
      int lane_t = (m & 15) | (((k0 >> 3) & 1) << 4);
      int e0 = (k0 >> 4) * 8;
      async_ld_b128(Ab + (size_t)(m0 + m) * K + kt * 32 + k0,
                    abuf + (m >> 4) * 1024 + lane_t * 32 + e0 * 2);
    }
#else
    #pragma unroll
    for (int i = 0; i < 2; ++i) {
      int idx = tid + i * 256;
      int m = idx >> 2, k0 = (idx & 3) * 8;
      st.ab[i] = *(const uint4*)(Ab + (size_t)(m0 + m) * K + kt * 32 + k0);
    }
#endif
  } else {
    #pragma unroll
    for (int i = 0; i < 4; ++i) {
      int idx = tid + i * 256;                 // 0..1023
      int m = idx >> 3, k0 = (idx & 7) * 4;    // float4 -> 4 bf16
      st.a4[i] = *(const float4*)(Af + (size_t)(m0 + m) * K + kt * 32 + k0);
    }
  }
  {
    int n = tid & 127, kq = tid >> 7;          // 16 k-rows per thread
    const float* wp = W + (size_t)(kt * 32 + kq * 8) * N + n0 + n;
    #pragma unroll
    for (int j = 0; j < 16; ++j) st.b[j] = wp[(size_t)((j & 7) + (j >> 3) * 16) * N];
  }
}

template<bool ABF16>
__device__ __forceinline__ void gemm_stage_store(
    const GemmStage& st, int tid, unsigned char* buf)
{
  if (ABF16) {
#if !HAVE_ASYNC_LDS
    #pragma unroll
    for (int i = 0; i < 2; ++i) {
      int idx = tid + i * 256;
      int m = idx >> 2, k0 = (idx & 3) * 8;
      int lane_t = (m & 15) | (((k0 >> 3) & 1) << 4);
      int e0 = (k0 >> 4) * 8;
      *(uint4*)(buf + (m >> 4) * 1024 + lane_t * 32 + e0 * 2) = st.ab[i];
    }
#endif
  } else {
    #pragma unroll
    for (int i = 0; i < 4; ++i) {
      int idx = tid + i * 256;
      int m = idx >> 3, k0 = (idx & 7) * 4;
      int krem = k0 & 15;
      int lane_t = (m & 15) | ((krem >> 3) << 4);
      int e0 = (k0 >> 4) * 8 + (krem & 7);
      uint2 pv; pv.x = pk2(st.a4[i].x, st.a4[i].y);
      pv.y = pk2(st.a4[i].z, st.a4[i].w);
      *(uint2*)(buf + (m >> 4) * 1024 + lane_t * 32 + e0 * 2) = pv;
    }
  }
  {
    int n = tid & 127, kq = tid >> 7;
    unsigned e[8];
    #pragma unroll
    for (int j = 0; j < 8; ++j) e[j] = pk2(st.b[2 * j], st.b[2 * j + 1]);
    unsigned char* dst = buf + 8192 + (n >> 4) * 1024 + ((n & 15) | (kq << 4)) * 32;
    *(uint4*)(dst)      = make_uint4(e[0], e[1], e[2], e[3]);
    *(uint4*)(dst + 16) = make_uint4(e[4], e[5], e[6], e[7]);
  }
}

template<bool ABF16, bool BIAS>
__global__ __launch_bounds__(256)
void gemm_k(const void* __restrict__ Aptr, const float* __restrict__ W,
            const float* __restrict__ bias, float* __restrict__ C,
            int M, int N, int K)
{
  __shared__ unsigned char sm[2][16384];       // per buffer: A [0,8K) B [8K,16K)
  const int tid  = threadIdx.x;
  const int wave = tid >> 5, lane = tid & 31;
  const int wrow = wave >> 2, wcol = wave & 3;
  const int hi   = lane >> 4, ln = lane & 15;
  const int m0 = blockIdx.y * 128, n0 = blockIdx.x * 128;

  f32x8 acc[4][2];
  #pragma unroll
  for (int i = 0; i < 4; ++i)
    #pragma unroll
    for (int j = 0; j < 2; ++j) acc[i][j] = zero8();

  const float*          Af = (const float*)Aptr;
  const unsigned short* Ab = (const unsigned short*)Aptr;
  const int KT = K >> 5;

  GemmStage st;
  gemm_stage_load<ABF16>(st, Af, Ab, W, K, N, m0, n0, 0, tid, sm[0]);
  gemm_stage_store<ABF16>(st, tid, sm[0]);
  if (ABF16) wait_async0();
  __syncthreads();

  for (int kt = 0; kt < KT; ++kt) {
    const int cur = kt & 1;
    unsigned char* smc = sm[cur];
    unsigned char* smn = sm[cur ^ 1];

    // issue next tile's global loads (and async LDS DMA) before the math
    if (kt + 1 < KT)
      gemm_stage_load<ABF16>(st, Af, Ab, W, K, N, m0, n0, kt + 1, tid, smn);
    if (kt + 2 < KT) {                          // global_prefetch_b8 tile kt+2
      const void* pa = ABF16
        ? (const void*)(Ab + (size_t)(m0 + (tid >> 1)) * K + (kt + 2) * 32)
        : (const void*)(Af + (size_t)(m0 + (tid >> 1)) * K + (kt + 2) * 32);
      __builtin_prefetch(pa, 0, 1);
      __builtin_prefetch(W + (size_t)((kt + 2) * 32 + (tid >> 3)) * N + n0, 0, 1);
    }

    // ---- 8 WMMAs per wave on the current buffer ----
    bf16x16 bfr[2];
    #pragma unroll
    for (int nt = 0; nt < 2; ++nt)
      bfr[nt] = ldfrag(smc + 8192 + (wcol * 2 + nt) * 1024 + lane * 32);
    #pragma unroll
    for (int mt = 0; mt < 4; ++mt) {
      bf16x16 afr = ldfrag(smc + (wrow * 4 + mt) * 1024 + lane * 32);
      acc[mt][0] = wmma_bf16(afr, bfr[0], acc[mt][0]);
      acc[mt][1] = wmma_bf16(afr, bfr[1], acc[mt][1]);
    }

    // convert + store next tile into the other buffer
    if (kt + 1 < KT) {
      gemm_stage_store<ABF16>(st, tid, smn);
      if (ABF16) wait_async0();
    }
    __syncthreads();                            // single barrier per K step
  }

  // ---- epilogue: C layout = lane n (ln), VGPR r -> row r + hi*8 ----
  #pragma unroll
  for (int mt = 0; mt < 4; ++mt)
    #pragma unroll
    for (int nt = 0; nt < 2; ++nt) {
      int col = n0 + wcol * 32 + nt * 16 + ln;
      float b = BIAS ? bias[col] : 0.f;
      #pragma unroll
      for (int r = 0; r < 8; ++r) {
        int row = m0 + wrow * 64 + mt * 16 + r + hi * 8;
        C[(size_t)row * N + col] = acc[mt][nt][r] + b;
      }
    }
}

// ---------------------------------------------------------------------------
// Fused RMSNorm + RoPE + layout. One block per (joint position, head),
// 128 threads = head_dim. Writes bf16 jq/jk ([h][s][128]) and transposed
// Vt ([h][128][s]) so the attention P*V B-fragments are contiguous.
// ---------------------------------------------------------------------------
__global__ __launch_bounds__(128)
void post_k(const float* __restrict__ qkv, const float* __restrict__ eqkv,
            const float* __restrict__ cosb, const float* __restrict__ sinb,
            const float* __restrict__ nqw, const float* __restrict__ nkw,
            const float* __restrict__ naqw, const float* __restrict__ nakw,
            unsigned short* __restrict__ jq, unsigned short* __restrict__ jk,
            unsigned short* __restrict__ vt)
{
  __shared__ float red[4];
  const int pos = blockIdx.x, h = blockIdx.y, d = threadIdx.x;
  const bool txt = pos < TXT;
  const float* src = txt ? (eqkv + (size_t)pos * N3)
                         : (qkv + (size_t)(pos - TXT) * N3);
  float q = src[h * HD + d];
  float k = src[INNER + h * HD + d];
  float v = src[2 * INNER + h * HD + d];

  float sq = q * q, sk = k * k;
  #pragma unroll
  for (int off = 16; off > 0; off >>= 1) {
    sq += __shfl_xor(sq, off, 32);
    sk += __shfl_xor(sk, off, 32);
  }
  const int wv = threadIdx.x >> 5;
  if ((threadIdx.x & 31) == 0) red[wv] = sq;
  __syncthreads();
  float tq = red[0] + red[1] + red[2] + red[3];
  __syncthreads();
  if ((threadIdx.x & 31) == 0) red[wv] = sk;
  __syncthreads();
  float tk = red[0] + red[1] + red[2] + red[3];

  float qn = q * rsqrtf(tq * (1.f / HD) + EPSV) * (txt ? naqw[d] : nqw[d]);
  float kn = k * rsqrtf(tk * (1.f / HD) + EPSV) * (txt ? nakw[d] : nkw[d]);

  float c = cosb[(size_t)pos * HD + d];
  float s = sinb[(size_t)pos * HD + d];
  float qp = __shfl_xor(qn, 1, 32);             // x[d^1]
  float kp = __shfl_xor(kn, 1, 32);
  float qr = (d & 1) ? qp : -qp;                // rot[2i]=-x[2i+1], rot[2i+1]=x[2i]
  float kr = (d & 1) ? kp : -kp;
  float qo = qn * c + qr * s;
  float ko = kn * c + kr * s;

  size_t o = ((size_t)h * SEQ + pos) * HD + d;
  jq[o] = f2bf(qo);
  jk[o] = f2bf(ko);
  vt[((size_t)h * HD + d) * SEQ + pos] = f2bf(v);
}

// ---------------------------------------------------------------------------
// Flash attention. Block = (head, 64 q rows), 4 waves, each wave owns 16 q
// rows. K-tile (128x128 bf16) and Vt-tile staged in LDS (async DMA when
// available); online softmax on 16x128 score registers; P written bf16 into
// an overlay of the K tile (after a barrier), then P@V.
// 64 WMMAs per wave per 128-key iteration.
// ---------------------------------------------------------------------------
__global__ __launch_bounds__(128)
void attn_k(const unsigned short* __restrict__ jq,
            const unsigned short* __restrict__ jk,
            const unsigned short* __restrict__ vt,
            unsigned short* __restrict__ aout)
{
  __shared__ unsigned char lds[65536];   // [0,32K): K tile (+P overlay)  [32K,64K): Vt
  unsigned char* ldsK = lds;
  unsigned char* ldsV = lds + 32768;

  const int tid = threadIdx.x, wave = tid >> 5, lane = tid & 31;
  const int hi = lane >> 4, ln = lane & 15;
  const int h = blockIdx.y, qt = blockIdx.x;
  const int qbase = qt * 64 + wave * 16;
  unsigned char* ldsP = ldsK + wave * 4096;     // 16 rows x 128 kpos bf16

  const unsigned short* jqh = jq + (size_t)h * SEQ * HD;
  const unsigned short* jkh = jk + (size_t)h * SEQ * HD;
  const unsigned short* vth = vt + (size_t)h * HD * SEQ;

  // Q A-fragments: 16 rows x full 128 head-dim = 4 chunks of K=32
  bf16x16 qf[4];
  {
    const unsigned short* qrow = jqh + (size_t)(qbase + ln) * HD;
    #pragma unroll
    for (int c = 0; c < 4; ++c)
      qf[c] = ldfrag(qrow + c * 32 + hi * 16);
  }

  f32x8 oacc[8];
  #pragma unroll
  for (int i = 0; i < 8; ++i) oacc[i] = zero8();
  float mi[8], li[8];
  #pragma unroll
  for (int r = 0; r < 8; ++r) { mi[r] = -__builtin_inff(); li[r] = 0.f; }

  for (int kb = 0; kb < SEQ; kb += 128) {
    __syncthreads();                       // prev iteration's LDS reads done
#if HAVE_ASYNC_LDS
    {
      const unsigned short* ksrc = jkh + (size_t)kb * HD;     // contiguous 32KB
      #pragma unroll
      for (int i = 0; i < 16; ++i)
        async_ld_b128(ksrc + (size_t)(tid + i * 128) * 8,
                      ldsK + (size_t)(tid + i * 128) * 16);
      const unsigned short* vsrc = vth + (size_t)tid * SEQ + kb;
      #pragma unroll
      for (int j = 0; j < 16; ++j)
        async_ld_b128(vsrc + (size_t)j * 8, ldsV + tid * 256 + j * 16);
      wait_async0();
    }
#else
    {
      const uint4* src = (const uint4*)(jkh + (size_t)kb * HD);
      uint4* dst = (uint4*)ldsK;
      #pragma unroll
      for (int i = 0; i < 16; ++i) dst[tid + i * 128] = src[tid + i * 128];
    }
    {
      const uint4* src = (const uint4*)(vth + (size_t)tid * SEQ + kb);
      uint4* dst = (uint4*)(ldsV + tid * 256);
      #pragma unroll
      for (int j = 0; j < 16; ++j) dst[j] = src[j];
    }
#endif
    __syncthreads();

    // ---- scores S = Q K^T : 8 col-tiles x 4 d-chunks ----
    f32x8 sacc[8];
    #pragma unroll
    for (int nt = 0; nt < 8; ++nt) sacc[nt] = zero8();
    #pragma unroll
    for (int nt = 0; nt < 8; ++nt)
      #pragma unroll
      for (int c = 0; c < 4; ++c) {
        bf16x16 bf = ldfrag(ldsK + (nt * 16 + ln) * 256 + (c * 32 + hi * 16) * 2);
        sacc[nt] = wmma_bf16(qf[c], bf, sacc[nt]);
      }
    __syncthreads();                       // all K-tile reads done before P overlay

    // ---- online softmax (row = 16-lane half of the wave) ----
    float mnew[8], alpha[8];
    #pragma unroll
    for (int r = 0; r < 8; ++r) {
      float mx = -__builtin_inff();
      #pragma unroll
      for (int nt = 0; nt < 8; ++nt) mx = fmaxf(mx, sacc[nt][r]);
      #pragma unroll
      for (int off = 1; off < 16; off <<= 1) mx = fmaxf(mx, __shfl_xor(mx, off, 32));
      mx *= SCALEV;
      mnew[r]  = fmaxf(mi[r], mx);
      alpha[r] = __expf(mi[r] - mnew[r]);
      mi[r]    = mnew[r];
    }
    #pragma unroll
    for (int nt = 0; nt < 8; ++nt)
      #pragma unroll
      for (int r = 0; r < 8; ++r)
        sacc[nt][r] = __expf(sacc[nt][r] * SCALEV - mnew[r]);
    #pragma unroll
    for (int r = 0; r < 8; ++r) {
      float rs = 0.f;
      #pragma unroll
      for (int nt = 0; nt < 8; ++nt) rs += sacc[nt][r];
      #pragma unroll
      for (int off = 1; off < 16; off <<= 1) rs += __shfl_xor(rs, off, 32);
      li[r] = li[r] * alpha[r] + rs;
      #pragma unroll
      for (int dt = 0; dt < 8; ++dt) oacc[dt][r] *= alpha[r];
    }

    // ---- P -> per-wave LDS (row-major 16x128 bf16) ----
    #pragma unroll
    for (int nt = 0; nt < 8; ++nt)
      #pragma unroll
      for (int r = 0; r < 8; ++r)
        *(unsigned short*)(ldsP + (r + hi * 8) * 256 + (nt * 16 + ln) * 2) =
            f2bf(sacc[nt][r]);

    // ---- O += P @ V ----
    #pragma unroll
    for (int c = 0; c < 4; ++c) {
      bf16x16 pf = ldfrag(ldsP + ln * 256 + (c * 32 + hi * 16) * 2);
      #pragma unroll
      for (int dt = 0; dt < 8; ++dt) {
        bf16x16 vf = ldfrag(ldsV + (dt * 16 + ln) * 256 + (c * 32 + hi * 16) * 2);
        oacc[dt] = wmma_bf16(pf, vf, oacc[dt]);
      }
    }
  }

  // ---- finalize: O / l, store bf16 into [seq][h*128+d] for the out-proj ----
  #pragma unroll
  for (int r = 0; r < 8; ++r) {
    float inv = 1.f / li[r];
    size_t row = qbase + r + hi * 8;
    #pragma unroll
    for (int dt = 0; dt < 8; ++dt)
      aout[row * INNER + h * HD + dt * 16 + ln] = f2bf(oacc[dt][r] * inv);
  }
}

// ---------------------------------------------------------------------------
// Workspace layout (bytes):
//   qkv  f32 [2048][6144]        @ 0          (50,331,648)
//   eqkv f32 [ 256][6144]        @ 50,331,648 ( 6,291,456)
//   jq   bf16 [16][2304][128]    @ 56,623,104 ( 9,437,184)
//   jk   bf16 [16][2304][128]    @ 66,060,288 ( 9,437,184)
//   vt   bf16 [16][128][2304]    @ 75,497,472 ( 9,437,184)
//   aout bf16 [2304][2048]       @ 84,934,656 ( 9,437,184)   total ~94.4 MB
// ---------------------------------------------------------------------------
extern "C" void kernel_launch(void* const* d_in, const int* in_sizes, int n_in,
                              void* d_out, int out_size, void* d_ws, size_t ws_size,
                              hipStream_t stream)
{
  (void)in_sizes; (void)n_in; (void)out_size; (void)ws_size;
  const float* hidden = (const float*)d_in[0];
  const float* enc    = (const float*)d_in[1];
  const float* fcos   = (const float*)d_in[2];
  const float* fsin   = (const float*)d_in[3];
  const float* Wqkv   = (const float*)d_in[4];
  const float* Wqkva  = (const float*)d_in[5];
  const float* bqkva  = (const float*)d_in[6];
  const float* nqw    = (const float*)d_in[7];
  const float* nkw    = (const float*)d_in[8];
  const float* naqw   = (const float*)d_in[9];
  const float* nakw   = (const float*)d_in[10];
  const float* Wout   = (const float*)d_in[11];
  const float* bout   = (const float*)d_in[12];
  const float* Wao    = (const float*)d_in[13];
  const float* bao    = (const float*)d_in[14];
  float* out = (float*)d_out;

  char* ws = (char*)d_ws;
  float*          qkv  = (float*)(ws);
  float*          eqkv = (float*)(ws + 50331648);
  unsigned short* jq   = (unsigned short*)(ws + 56623104);
  unsigned short* jk   = (unsigned short*)(ws + 66060288);
  unsigned short* vt   = (unsigned short*)(ws + 75497472);
  unsigned short* ao   = (unsigned short*)(ws + 84934656);

  // QKV projections (A fp32 -> bf16 staging)
  gemm_k<false,false><<<dim3(N3/128, IMG/128), 256, 0, stream>>>(
      hidden, Wqkv, nullptr, qkv, IMG, N3, DM);
  gemm_k<false,true ><<<dim3(N3/128, TXT/128), 256, 0, stream>>>(
      enc, Wqkva, bqkva, eqkv, TXT, N3, DM);

  // RMSNorm + RoPE + bf16 relayout
  post_k<<<dim3(SEQ, HEADS), 128, 0, stream>>>(
      qkv, eqkv, fcos, fsin, nqw, nkw, naqw, nakw, jq, jk, vt);

  // Flash attention over the joint sequence
  attn_k<<<dim3(SEQ/64, HEADS), 128, 0, stream>>>(jq, jk, vt, ao);

  // Output projections: img rows TXT..end -> img_out, txt rows -> enc_out
  gemm_k<true,true><<<dim3(DM/128, IMG/128), 256, 0, stream>>>(
      ao + (size_t)TXT * INNER, Wout, bout, out, IMG, DM, INNER);
  gemm_k<true,true><<<dim3(DM/128, TXT/128), 256, 0, stream>>>(
      ao, Wao, bao, out + (size_t)IMG * DM, TXT, DM, INNER);
}
